// AttentionBlock_18064632447309
// MI455X (gfx1250) — matrix operand order repaired and checked by reading, not compiled, not run on hardware
//
#include <hip/hip_runtime.h>
#include <hip/hip_bf16.h>

// ---------------------------------------------------------------------------
// MI455X (gfx1250) transformer block: LN -> QKV -> flash-attention -> LN ->
// fused MLP, all GEMMs via v_wmma_f32_16x16x32_f16 (wave32, 8 waves/block).
// ---------------------------------------------------------------------------

typedef __attribute__((ext_vector_type(16))) _Float16 v16h;
typedef __attribute__((ext_vector_type(8)))  float    v8f;

#define D_MODEL 256
#define MLP_DIM 1024
#define SEQ     4096                 // n*h tokens per batch
#define BATCH   8
#define TOKENS  (BATCH * SEQ)        // 32768

// ---- WMMA fragment helpers (per CDNA5 ISA 7.12.2 VGPR layouts) -------------

// A matrix 16x32 f16, source row-major with leading dim `ld`.
// lane(0..31): half = lane>>4, m = lane&15 ; pair p: K = 8*half + 2*(p&3) + 16*(p>>2)
__device__ inline v16h frag_a(const _Float16* __restrict__ a, int ld) {
  const int lane = threadIdx.x & 31;
  const int half = lane >> 4, m = lane & 15;
  const _Float16* row = a + m * ld;
  v16h f;
#pragma unroll
  for (int p = 0; p < 8; ++p) {
    const int k = 8 * half + ((p & 3) << 1) + ((p >> 2) << 4);
    f[2 * p]     = row[k];
    f[2 * p + 1] = row[k + 1];
  }
  return f;
}

// B matrix 32x16 f16 where B[k][n] = src[n][k] (src row-major, leading dim ld).
// lane: half = lane>>4, n = lane&15 ; pair p: K = 8*half + 2*(p&3) + 16*(p>>2)
__device__ inline v16h frag_bt(const _Float16* __restrict__ src, int ld) {
  const int lane = threadIdx.x & 31;
  const int half = lane >> 4, n = lane & 15;
  const _Float16* col = src + (size_t)n * ld;
  v16h f;
#pragma unroll
  for (int p = 0; p < 8; ++p) {
    const int k = 8 * half + ((p & 3) << 1) + ((p >> 2) << 4);
    f[2 * p]     = col[k];
    f[2 * p + 1] = col[k + 1];
  }
  return f;
}

__device__ inline v8f wmma(v16h a, v16h b, v8f c) {
  return __builtin_amdgcn_wmma_f32_16x16x32_f16(false, a, false, b,
                                                (short)0, c, false, false);
}

// ---- small utility kernels -------------------------------------------------

__global__ void k_cvt_f16(const float* __restrict__ src,
                          _Float16* __restrict__ dst, int n) {
  int i = blockIdx.x * blockDim.x + threadIdx.x;
  if (i < n) dst[i] = (_Float16)src[i];
}

// one wave32 per 256-wide token row
__global__ void k_layernorm(const float* __restrict__ x,
                            const float* __restrict__ g,
                            const float* __restrict__ bta,
                            _Float16* __restrict__ out) {
  const int lane = threadIdx.x & 31;
  const int wave = threadIdx.x >> 5;
  const int t = blockIdx.x * 8 + wave;
  const float* row = x + (size_t)t * D_MODEL;
  float v[8], s = 0.f, ss = 0.f;
#pragma unroll
  for (int q = 0; q < 8; ++q) {
    v[q] = row[lane + 32 * q];
    s += v[q]; ss += v[q] * v[q];
  }
#pragma unroll
  for (int off = 16; off > 0; off >>= 1) {
    s  += __shfl_xor(s,  off, 32);
    ss += __shfl_xor(ss, off, 32);
  }
  const float mu  = s * (1.f / D_MODEL);
  const float var = ss * (1.f / D_MODEL) - mu * mu;
  const float rs  = rsqrtf(var + 1e-5f);
#pragma unroll
  for (int q = 0; q < 8; ++q) {
    const int c = lane + 32 * q;
    out[(size_t)t * D_MODEL + c] = (_Float16)((v[q] - mu) * rs * g[c] + bta[c]);
  }
}

// ---- QKV projection: C[32768,768] = xn * Wqkv^T, de-interleave q/k/vT ------

__global__ void k_qkv(const _Float16* __restrict__ xn,
                      const _Float16* __restrict__ wq,
                      _Float16* __restrict__ q,
                      _Float16* __restrict__ kk,
                      _Float16* __restrict__ vt) {
  __shared__ _Float16 xs[16 * D_MODEL];       // 8 KB A tile, shared by 8 waves
  const int t0 = blockIdx.x * 16;
  const int e0 = blockIdx.y * 128;
  const int wave = threadIdx.x >> 5;
  for (int i = threadIdx.x; i < 16 * D_MODEL; i += blockDim.x)
    xs[i] = xn[(size_t)(t0 + (i >> 8)) * D_MODEL + (i & 255)];
  __syncthreads();

  const int ew = e0 + wave * 16;
  v8f acc = {};
#pragma unroll
  for (int ks = 0; ks < 8; ++ks) {
    if (ks < 7) __builtin_prefetch(wq + (size_t)ew * D_MODEL + (ks + 1) * 32, 0, 0);
    v16h a = frag_a(xs + ks * 32, D_MODEL);
    v16h b = frag_bt(wq + (size_t)ew * D_MODEL + ks * 32, D_MODEL);
    acc = wmma(a, b, acc);
  }
  const int lane = threadIdx.x & 31;
  const int half = lane >> 4, n = lane & 15;
  const int e = ew + n;                       // column in 0..767
  const int j = e % 3, ii = e / 3;            // torch reshape (d,3)
#pragma unroll
  for (int v = 0; v < 8; ++v) {
    const int row = t0 + v + 8 * half;
    const _Float16 val = (_Float16)acc[v];
    if (j == 0)      q [(size_t)row * D_MODEL + ii] = val;
    else if (j == 1) kk[(size_t)row * D_MODEL + ii] = val;
    else {
      const int b_ = row >> 12, tt = row & (SEQ - 1);
      vt[((size_t)(b_ * D_MODEL + ii) << 12) + tt] = val;   // vT[b][d][t]
    }
  }
}

// ---- flash attention: 16 query rows per block, 128-key slabs ---------------

__global__ void k_attn(const _Float16* __restrict__ q,
                       const _Float16* __restrict__ kk,
                       const _Float16* __restrict__ vt,
                       const float* __restrict__ x,
                       float* __restrict__ x1) {
  __shared__ _Float16 qs[16 * D_MODEL];   // 8 KB  (Q pre-scaled by d^-1/2)
  __shared__ float    sS[16 * 128];       // 8 KB  score slab
  __shared__ _Float16 sP[16 * 128];       // 4 KB  exp(S - m)
  __shared__ float sM[16], sL[16], sA[16];

  const int b   = blockIdx.y;
  const int t0  = blockIdx.x * 16;
  const int wave = threadIdx.x >> 5;
  const int lane = threadIdx.x & 31;
  const int half = lane >> 4, n16 = lane & 15;
  const float scale = 0.0625f;            // 256^-0.5

  const size_t qbase = ((size_t)b * SEQ + t0) * D_MODEL;
  for (int i = threadIdx.x; i < 16 * D_MODEL; i += blockDim.x)
    qs[i] = (_Float16)((float)q[qbase + (size_t)(i >> 8) * D_MODEL + (i & 255)] * scale);
  if (threadIdx.x < 16) { sM[threadIdx.x] = -3.0e38f; sL[threadIdx.x] = 0.f; }
  __syncthreads();

  v8f o0 = {}, o1 = {};
  const _Float16* kbat = kk + (size_t)b * SEQ * D_MODEL;
  const _Float16* vbat = vt + (size_t)b * D_MODEL * SEQ;
  const int c0 = wave * 32, c1 = c0 + 16;

  for (int kv0 = 0; kv0 < SEQ; kv0 += 128) {
    // S slab: wave w covers keys [kv0 + 16w, +16)
    const int n0 = kv0 + wave * 16;
    v8f s = {};
#pragma unroll
    for (int ks = 0; ks < 8; ++ks) {
      v16h a  = frag_a(qs + ks * 32, D_MODEL);
      v16h bb = frag_bt(kbat + (size_t)n0 * D_MODEL + ks * 32, D_MODEL);
      s = wmma(a, bb, s);
    }
#pragma unroll
    for (int v = 0; v < 8; ++v)
      sS[(v + 8 * half) * 128 + wave * 16 + n16] = s[v];
    __syncthreads();

    // online softmax row state (16 rows handled by threads 0..15)
    if (threadIdx.x < 16) {
      const int r = threadIdx.x;
      float mold = sM[r], mx = mold;
      for (int c = 0; c < 128; ++c) mx = fmaxf(mx, sS[r * 128 + c]);
      const float alpha = __expf(mold - mx);
      float l = sL[r] * alpha;
      for (int c = 0; c < 128; ++c) {
        const float p = __expf(sS[r * 128 + c] - mx);
        l += p;
        sP[r * 128 + c] = (_Float16)p;
      }
      sM[r] = mx; sL[r] = l; sA[r] = alpha;
    }
    __syncthreads();

    // rescale running O and accumulate P * V (wave w owns dims [32w, 32w+32))
#pragma unroll
    for (int v = 0; v < 8; ++v) {
      const float al = sA[v + 8 * half];
      o0[v] *= al; o1[v] *= al;
    }
#pragma unroll
    for (int ks = 0; ks < 4; ++ks) {
      v16h a  = frag_a(sP + ks * 32, 128);
      v16h b0 = frag_bt(vbat + (size_t)c0 * SEQ + kv0 + ks * 32, SEQ);
      v16h b1 = frag_bt(vbat + (size_t)c1 * SEQ + kv0 + ks * 32, SEQ);
      o0 = wmma(a, b0, o0);
      o1 = wmma(a, b1, o1);
    }
    __syncthreads();
  }

  // normalize, add residual, store f32
#pragma unroll
  for (int v = 0; v < 8; ++v) {
    const int r = v + 8 * half;
    const float inv = 1.0f / sL[r];
    const size_t i0 = ((size_t)b * SEQ + t0 + r) * D_MODEL + c0 + n16;
    x1[i0]      = x[i0]      + o0[v] * inv;
    x1[i0 + 16] = x[i0 + 16] + o1[v] * inv;
  }
}

// ---- fused MLP: hidden(16x1024) streamed through LDS in 128-wide chunks ----

__global__ void k_mlp(const _Float16* __restrict__ xn2,
                      const _Float16* __restrict__ w1f,
                      const float* __restrict__ b1,
                      const _Float16* __restrict__ w2f,
                      const float* __restrict__ b2,
                      const float* __restrict__ x1,
                      float* __restrict__ out) {
  __shared__ _Float16 xs[16 * D_MODEL];   // 8 KB LN'd tokens
  __shared__ _Float16 hs[16 * 128];       // 4 KB SiLU(hidden) chunk
  const int t0 = blockIdx.x * 16;
  const int wave = threadIdx.x >> 5;
  const int lane = threadIdx.x & 31;
  const int half = lane >> 4, n16 = lane & 15;
  for (int i = threadIdx.x; i < 16 * D_MODEL; i += blockDim.x)
    xs[i] = xn2[(size_t)(t0 + (i >> 8)) * D_MODEL + (i & 255)];
  __syncthreads();

  v8f o0 = {}, o1 = {};
  const int c0 = wave * 32, c1 = c0 + 16;
  for (int mc = 0; mc < MLP_DIM; mc += 128) {
    const int m0 = mc + wave * 16;
    v8f h = {};
#pragma unroll
    for (int ks = 0; ks < 8; ++ks) {
      if (ks < 7) __builtin_prefetch(w1f + (size_t)m0 * D_MODEL + (ks + 1) * 32, 0, 0);
      v16h a  = frag_a(xs + ks * 32, D_MODEL);
      v16h bb = frag_bt(w1f + (size_t)m0 * D_MODEL + ks * 32, D_MODEL);
      h = wmma(a, bb, h);
    }
    const float bias = b1[m0 + n16];
#pragma unroll
    for (int v = 0; v < 8; ++v) {
      float hv = h[v] + bias;
      hv = hv / (1.f + __expf(-hv));                   // SiLU
      hs[(v + 8 * half) * 128 + wave * 16 + n16] = (_Float16)hv;
    }
    __syncthreads();
#pragma unroll
    for (int ks = 0; ks < 4; ++ks) {
      v16h a   = frag_a(hs + ks * 32, 128);
      v16h bb0 = frag_bt(w2f + (size_t)c0 * MLP_DIM + mc + ks * 32, MLP_DIM);
      v16h bb1 = frag_bt(w2f + (size_t)c1 * MLP_DIM + mc + ks * 32, MLP_DIM);
      o0 = wmma(a, bb0, o0);
      o1 = wmma(a, bb1, o1);
    }
    __syncthreads();
  }
#pragma unroll
  for (int v = 0; v < 8; ++v) {
    const size_t i0 = (size_t)(t0 + v + 8 * half) * D_MODEL + c0 + n16;
    out[i0]      = x1[i0]      + o0[v] + b2[c0 + n16];
    out[i0 + 16] = x1[i0 + 16] + o1[v] + b2[c0 + 16 + n16];
  }
}

// ---------------------------------------------------------------------------

extern "C" void kernel_launch(void* const* d_in, const int* in_sizes, int n_in,
                              void* d_out, int out_size, void* d_ws, size_t ws_size,
                              hipStream_t stream) {
  const float* x     = (const float*)d_in[0];
  const float* w_qkv = (const float*)d_in[1];
  const float* g1    = (const float*)d_in[2];
  const float* be1   = (const float*)d_in[3];
  const float* g2    = (const float*)d_in[4];
  const float* be2   = (const float*)d_in[5];
  const float* w1    = (const float*)d_in[6];
  const float* b1    = (const float*)d_in[7];
  const float* w2    = (const float*)d_in[8];
  const float* b2    = (const float*)d_in[9];
  float* out = (float*)d_out;

  char* ws = (char*)d_ws;
  const size_t MB = (size_t)1 << 20;
  _Float16* xn  = (_Float16*)(ws);                 // 16 MB (reused for xn2)
  _Float16* q   = (_Float16*)(ws + 16 * MB);       // 16 MB
  _Float16* kk  = (_Float16*)(ws + 32 * MB);       // 16 MB
  _Float16* vt  = (_Float16*)(ws + 48 * MB);       // 16 MB (transposed V)
  float*    x1  = (float*)   (ws + 64 * MB);       // 32 MB residual stream
  _Float16* wqf = (_Float16*)(ws + 96 * MB);       // f16 weights (~1.4 MB)
  _Float16* w1f = wqf + 768 * 256;
  _Float16* w2f = w1f + 1024 * 256;

  // weights -> f16 (stay L2-resident afterwards)
  k_cvt_f16<<<(768 * 256 + 255) / 256, 256, 0, stream>>>(w_qkv, wqf, 768 * 256);
  k_cvt_f16<<<(1024 * 256 + 255) / 256, 256, 0, stream>>>(w1, w1f, 1024 * 256);
  k_cvt_f16<<<(256 * 1024 + 255) / 256, 256, 0, stream>>>(w2, w2f, 256 * 1024);

  k_layernorm<<<TOKENS / 8, 256, 0, stream>>>(x, g1, be1, xn);
  k_qkv<<<dim3(TOKENS / 16, 768 / 128), 256, 0, stream>>>(xn, wqf, q, kk, vt);
  k_attn<<<dim3(SEQ / 16, BATCH), 256, 0, stream>>>(q, kk, vt, x, x1);
  k_layernorm<<<TOKENS / 8, 256, 0, stream>>>(x1, g2, be2, xn);
  k_mlp<<<TOKENS / 16, 256, 0, stream>>>(xn, w1f, b1, w2f, b2, x1, out);
}
